// GPT3Attention_41961830481966
// MI455X (gfx1250) — matrix operand order repaired and checked by reading, not compiled, not run on hardware
//
#include <hip/hip_runtime.h>
#include <stdint.h>

#define Bsz 2
#define Sq  2048
#define Dm  2048
#define Hn  16
#define DP  128
#define Mrows (Bsz * Sq)                 // 4096
#define SCALE 0.08838834764831845f      // 1/sqrt(128)

typedef __attribute__((ext_vector_type(16))) __bf16 v16bf;
typedef __attribute__((ext_vector_type(8)))  __bf16 v8bf;
typedef __attribute__((ext_vector_type(8)))  float  v8f;
typedef __attribute__((ext_vector_type(4)))  uint32_t u32x4;
typedef __attribute__((ext_vector_type(8)))  uint32_t u32x8;

union ABFrag { v16bf v; v8bf h[2]; };

// fp32 -> bf16 round-to-nearest-even
static __device__ __forceinline__ __bf16 f2bf(float f) {
  union { float f; uint32_t u; } x; x.f = f;
  uint32_t r = x.u + 0x7FFFu + ((x.u >> 16) & 1u);
  uint16_t hb = (uint16_t)(r >> 16);
  return __builtin_bit_cast(__bf16, hb);
}

// stage 8 contiguous elements into LDS, converting if needed
static __device__ __forceinline__ void stage8(const float* __restrict__ g, __bf16* s) {
#pragma unroll
  for (int i = 0; i < 8; ++i) s[i] = f2bf(g[i]);
}
static __device__ __forceinline__ void stage8(const __bf16* __restrict__ g, __bf16* s) {
  *(v8bf*)s = *(const v8bf*)g;
}

// ---------------------------------------------------------------------------
// Tensor Data Mover: 2D tile load global->LDS (D# groups per ISA 08 §8).
//   dim0    : tensor extent along contiguous axis (elements, for OOB)
//   dim1    : tensor extent along row axis
//   tile0   : tile width  (elements, contiguous)
//   tile1   : tile height (rows)
//   stride0 : row stride in elements
//   w0flags : data_size / pad_enable / pad_interval / pad_amount bits
// ---------------------------------------------------------------------------
static __device__ __forceinline__ void tdm_load_2d(
    const void* gptr, uint32_t lds_off, uint32_t w0flags,
    uint32_t dim0, uint32_t dim1, uint32_t tile0, uint32_t tile1,
    uint32_t stride0) {
  uint64_t ga = (uint64_t)(uintptr_t)gptr;
  u32x4 g0;
  g0[0] = 1u;                                            // count = 1 valid D#
  g0[1] = lds_off;                                       // LDS byte address
  g0[2] = (uint32_t)ga;                                  // global_addr[31:0]
  g0[3] = (uint32_t)((ga >> 32) & 0x01FFFFFFu) | 0x80000000u;  // addr[56:32] | type=2
  u32x8 g1;
  g1[0] = w0flags;                                       // mask=0 | data_size | pad ctl
  g1[1] = (dim0 & 0xFFFFu) << 16;                        // tensor_dim0[15:0]
  g1[2] = (dim0 >> 16) | ((dim1 & 0xFFFFu) << 16);       // dim0[31:16] | dim1[15:0]
  g1[3] = (dim1 >> 16) | (tile0 << 16);                  // dim1[31:16] | tile_dim0
  g1[4] = tile1;                                         // tile_dim1 | tile_dim2=0
  g1[5] = stride0;                                       // tensor_dim0_stride[31:0]
  g1[6] = 0u;                                            // stride0[47:32] | stride1 lo
  g1[7] = 0u;                                            // stride1 hi
  asm volatile("tensor_load_to_lds %0, %1" : : "s"(g0), "s"(g1) : "memory");
}

// data_size=1 (2B) | pad_enable | pad_interval | pad_amount(=4 DWORDs -> enc 3)
#define TDM_W0_K ((1u << 16) | (1u << 20) | (5u << 22) | (3u << 25))  // 256B rows
#define TDM_W0_V ((1u << 16) | (1u << 20) | (3u << 22) | (3u << 25))  // 64B rows

// ---------------------------------------------------------------------------
// GEMM: out[m,n] = A[m,:] . W[n,:] + bias[n]   (y = x @ W^T + b, "NT" gemm)
// Block = 128x128 tile, 256 threads = 8 waves in 2(m) x 4(n); wave = 64x32.
// OUT_MODE 0: bf16 output in head-split layout [B,H,S,128]
// OUT_MODE 2: fp32 output, plain row-major [M, Dm]
// ---------------------------------------------------------------------------
template <typename TA, typename TB, int OUT_MODE>
__global__ __launch_bounds__(256) void gemm_nt_wmma(
    const TA* __restrict__ A, const TB* __restrict__ W,
    const float* __restrict__ bias, void* __restrict__ outp, int K) {
  __shared__ __bf16 As[128 * 40];   // padded row stride 40 (bank-conflict free)
  __shared__ __bf16 Ws[128 * 40];

  const int tid  = threadIdx.x;
  const int lane = tid & 31;
  const int wid  = tid >> 5;
  const int wm   = wid & 1;
  const int wn   = wid >> 1;
  const int r    = lane & 15;
  const int hi   = lane >> 4;
  const int m0   = blockIdx.y * 128;
  const int n0   = blockIdx.x * 128;

  v8f acc[4][2];
#pragma unroll
  for (int i = 0; i < 4; ++i)
#pragma unroll
    for (int j = 0; j < 2; ++j)
#pragma unroll
      for (int e = 0; e < 8; ++e) acc[i][j][e] = 0.f;

  for (int k0 = 0; k0 < K; k0 += 32) {
#pragma unroll
    for (int i = tid; i < 512; i += 256) {     // 512 16B-chunks per tile
      int row = i >> 2, cc = (i & 3) * 8;
      stage8(A + (size_t)(m0 + row) * K + k0 + cc, &As[row * 40 + cc]);
      stage8(W + (size_t)(n0 + row) * K + k0 + cc, &Ws[row * 40 + cc]);
    }
    __syncthreads();

    ABFrag af[4], bf[2];
#pragma unroll
    for (int i = 0; i < 4; ++i) {              // A 16x32: K {0..7|8..15} and {16..23|24..31}
      const __bf16* p = &As[(wm * 64 + i * 16 + r) * 40];
      af[i].h[0] = *(const v8bf*)(p + hi * 8);
      af[i].h[1] = *(const v8bf*)(p + 16 + hi * 8);
    }
#pragma unroll
    for (int j = 0; j < 2; ++j) {              // B 32x16: lane=n, K {0..7|8..15} and {16..23|24..31}
      const __bf16* p = &Ws[(wn * 32 + j * 16 + r) * 40];
      bf[j].h[0] = *(const v8bf*)(p + hi * 8);
      bf[j].h[1] = *(const v8bf*)(p + 16 + hi * 8);
    }
#pragma unroll
    for (int i = 0; i < 4; ++i)
#pragma unroll
      for (int j = 0; j < 2; ++j)
        acc[i][j] = __builtin_amdgcn_wmma_f32_16x16x32_bf16(
            false, af[i].v, false, bf[j].v, (short)0, acc[i][j], false, false);
    __syncthreads();
  }

#pragma unroll
  for (int i = 0; i < 4; ++i) {
#pragma unroll
    for (int j = 0; j < 2; ++j) {
      int ng = n0 + wn * 32 + j * 16 + r;
      float bv = bias[ng];
#pragma unroll
      for (int e = 0; e < 8; ++e) {
        int mg = m0 + wm * 64 + i * 16 + e + 8 * hi;
        float val = acc[i][j][e] + bv;
        if constexpr (OUT_MODE == 0) {
          int bb = mg >> 11, ss = mg & (Sq - 1);
          int hh = ng >> 7,  dd = ng & (DP - 1);
          ((__bf16*)outp)[((((size_t)bb * Hn + hh) * Sq + ss) << 7) + dd] = f2bf(val);
        } else {
          ((float*)outp)[(size_t)mg * Dm + ng] = val;
        }
      }
    }
  }
}

// ---------------------------------------------------------------------------
// Vh [B,H,S,128] -> VhT [B,H,128,S]  (bf16, LDS 32x32 tiles)
// ---------------------------------------------------------------------------
__global__ __launch_bounds__(256) void transpose_v_kernel(
    const __bf16* __restrict__ Vh, __bf16* __restrict__ VhT) {
  __shared__ __bf16 tile[32][33];
  int s0 = blockIdx.x * 32, d0 = blockIdx.y * 32;
  size_t base  = (size_t)blockIdx.z * Sq * DP;
  size_t baseT = (size_t)blockIdx.z * DP * Sq;
#pragma unroll
  for (int i = threadIdx.y; i < 32; i += 8)
    tile[i][threadIdx.x] = Vh[base + (size_t)(s0 + i) * DP + d0 + threadIdx.x];
  __syncthreads();
#pragma unroll
  for (int i = threadIdx.y; i < 32; i += 8)
    VhT[baseT + (size_t)(d0 + i) * Sq + s0 + threadIdx.x] = tile[threadIdx.x][i];
}

// ---------------------------------------------------------------------------
// Causal flash attention: 1 wave handles 16 query rows of one (b,h).
// K/V tiles double-buffered into LDS by the Tensor Data Mover; the next
// iteration's DMA is issued before waiting on the current one (TENSORcnt).
// Fragment reads come from TDM-padded LDS rows (K: 272B, V: 80B strides).
// ---------------------------------------------------------------------------
__global__ __launch_bounds__(32) void flash_attn_kernel(
    const __bf16* __restrict__ Qh, const __bf16* __restrict__ Kh,
    const __bf16* __restrict__ VhT, __bf16* __restrict__ Oh) {
  const int qt = blockIdx.x, h = blockIdx.y, b = blockIdx.z;
  const int lane = threadIdx.x & 31;
  const int r = lane & 15, hi = lane >> 4;

  const __bf16* Qb = Qh + ((size_t)(b * Hn + h) * Sq + qt * 16) * DP;
  const __bf16* Kb = Kh + (size_t)(b * Hn + h) * Sq * DP;
  const __bf16* Vb = VhT + (size_t)(b * Hn + h) * DP * Sq;

  __shared__ __bf16 Kl[2][32 * 136];   // 32 rows x (256B + 16B TDM pad) = 272B
  __shared__ __bf16 Vl[2][128 * 40];   // 128 rows x (64B + 16B TDM pad) = 80B
  __shared__ __bf16 pS[16 * 40];       // probs re-layout buffer (padded)

  const uint32_t ldsK[2] = { (uint32_t)(uintptr_t)&Kl[0][0], (uint32_t)(uintptr_t)&Kl[1][0] };
  const uint32_t ldsV[2] = { (uint32_t)(uintptr_t)&Vl[0][0], (uint32_t)(uintptr_t)&Vl[1][0] };

  ABFrag qf[4];                        // Q tile 16x128 as 4 A-fragments
#pragma unroll
  for (int dk = 0; dk < 4; ++dk) {
    const __bf16* p = Qb + (size_t)r * DP + dk * 32;
    qf[dk].h[0] = *(const v8bf*)(p + hi * 8);
    qf[dk].h[1] = *(const v8bf*)(p + 16 + hi * 8);
  }

  v8f o[8];
  float mrow[8], lrow[8];
#pragma unroll
  for (int e = 0; e < 8; ++e) { mrow[e] = -3.0e38f; lrow[e] = 0.f; }
#pragma unroll
  for (int dt = 0; dt < 8; ++dt)
#pragma unroll
    for (int e = 0; e < 8; ++e) o[dt][e] = 0.f;

  const int nkt = (qt >> 1) + 1;       // causal: only key tiles with keys <= q

  // TDM prologue: tiles for kt = 0 into buffer 0
  tdm_load_2d(Kb, ldsK[0], TDM_W0_K, DP, Sq, DP, 32, DP);
  tdm_load_2d(Vb, ldsV[0], TDM_W0_V, Sq, DP, 32, DP, Sq);

  for (int kt = 0; kt < nkt; ++kt) {
    const int t0 = kt * 32;
    const int cur = kt & 1;
    if (kt + 1 < nkt) {                // prefetch next tiles into other buffer
      const int t1 = t0 + 32, nb = cur ^ 1;
      tdm_load_2d(Kb + (size_t)t1 * DP, ldsK[nb], TDM_W0_K, DP, Sq - t1, DP, 32, DP);
      tdm_load_2d(Vb + t1, ldsV[nb], TDM_W0_V, Sq - t1, DP, 32, DP, Sq);
      __builtin_amdgcn_s_wait_tensorcnt(2);  // current pair complete
    } else {
      __builtin_amdgcn_s_wait_tensorcnt(0);
    }

    v8f sc[2];
#pragma unroll
    for (int nt = 0; nt < 2; ++nt) {   // S = Q . K^T over depth=128
      v8f s;
#pragma unroll
      for (int e = 0; e < 8; ++e) s[e] = 0.f;
#pragma unroll
      for (int dk = 0; dk < 4; ++dk) {
        ABFrag kf;                     // B-frag: lane = key, depth contiguous
        const __bf16* p = &Kl[cur][(nt * 16 + r) * 136 + dk * 32 + hi * 8];
        kf.h[0] = *(const v8bf*)p;
        kf.h[1] = *(const v8bf*)(p + 16);
        s = __builtin_amdgcn_wmma_f32_16x16x32_bf16(
            false, qf[dk].v, false, kf.v, (short)0, s, false, false);
      }
      const int kg  = t0 + nt * 16 + r;    // global key index (this lane's column)
      const int qg0 = qt * 16 + 8 * hi;    // global query row of element e=0
#pragma unroll
      for (int e = 0; e < 8; ++e) {
        float v = s[e] * SCALE;
        if (kg > qg0 + e) v -= 1.0e9f;     // additive causal mask (matches reference)
        s[e] = v;
      }
      sc[nt] = s;
    }

    // row max over 32 keys: rows live in one 16-lane half -> xor masks 1,2,4,8
    float tmax[8], psum[8], alpha[8];
#pragma unroll
    for (int e = 0; e < 8; ++e) tmax[e] = fmaxf(sc[0][e], sc[1][e]);
#pragma unroll
    for (int msk = 1; msk < 16; msk <<= 1)
#pragma unroll
      for (int e = 0; e < 8; ++e) tmax[e] = fmaxf(tmax[e], __shfl_xor(tmax[e], msk, 32));
#pragma unroll
    for (int e = 0; e < 8; ++e) {
      float mnew = fmaxf(mrow[e], tmax[e]);
      alpha[e] = __expf(mrow[e] - mnew);
      mrow[e] = mnew;
      psum[e] = 0.f;
    }

    __syncthreads();                   // WAR guard on pS vs previous iteration
#pragma unroll
    for (int nt = 0; nt < 2; ++nt)
#pragma unroll
      for (int e = 0; e < 8; ++e) {
        float p = __expf(sc[nt][e] - mrow[e]);
        psum[e] += p;
        pS[(e + 8 * hi) * 40 + nt * 16 + r] = f2bf(p);
      }
#pragma unroll
    for (int msk = 1; msk < 16; msk <<= 1)
#pragma unroll
      for (int e = 0; e < 8; ++e) psum[e] += __shfl_xor(psum[e], msk, 32);
#pragma unroll
    for (int e = 0; e < 8; ++e) lrow[e] = lrow[e] * alpha[e] + psum[e];
#pragma unroll
    for (int dt = 0; dt < 8; ++dt)
#pragma unroll
      for (int e = 0; e < 8; ++e) o[dt][e] *= alpha[e];
    __syncthreads();                   // pS visible for A-fragment reads

    ABFrag pf;                         // P 16x32 as A-fragment
    pf.h[0] = *(const v8bf*)&pS[r * 40 + hi * 8];
    pf.h[1] = *(const v8bf*)&pS[r * 40 + 16 + hi * 8];
#pragma unroll
    for (int dt = 0; dt < 8; ++dt) {   // O += P . V  (V^T rows contiguous in t)
      ABFrag vf;
      const __bf16* p = &Vl[cur][(dt * 16 + r) * 40 + hi * 8];
      vf.h[0] = *(const v8bf*)p;
      vf.h[1] = *(const v8bf*)(p + 16);
      o[dt] = __builtin_amdgcn_wmma_f32_16x16x32_bf16(
          false, pf.v, false, vf.v, (short)0, o[dt], false, false);
    }
  }

#pragma unroll
  for (int e = 0; e < 8; ++e) lrow[e] = 1.f / lrow[e];
#pragma unroll
  for (int dt = 0; dt < 8; ++dt)
#pragma unroll
    for (int e = 0; e < 8; ++e) {
      int sg = qt * 16 + e + 8 * hi;   // merged-head layout: out[b, s, h*128 + d]
      Oh[((size_t)b * Sq + sg) * Dm + h * DP + dt * 16 + r] = f2bf(o[dt][e] * lrow[e]);
    }
}

// ---------------------------------------------------------------------------
extern "C" void kernel_launch(void* const* d_in, const int* in_sizes, int n_in,
                              void* d_out, int out_size, void* d_ws, size_t ws_size,
                              hipStream_t stream) {
  (void)in_sizes; (void)n_in; (void)out_size; (void)ws_size;
  const float* q    = (const float*)d_in[0];
  const float* k    = (const float*)d_in[1];
  const float* v    = (const float*)d_in[2];
  // d_in[3] = causal mask; applied analytically inside flash_attn_kernel
  const float* wq_w = (const float*)d_in[4];
  const float* wq_b = (const float*)d_in[5];
  const float* wk_w = (const float*)d_in[6];
  const float* wk_b = (const float*)d_in[7];
  const float* wv_w = (const float*)d_in[8];
  const float* wv_b = (const float*)d_in[9];
  const float* wd_w = (const float*)d_in[10];
  const float* wd_b = (const float*)d_in[11];

  const size_t nElem = (size_t)Bsz * Sq * Dm;   // 8,388,608 elements / buffer
  char* w = (char*)d_ws;                        // 5 bf16 buffers = ~84 MB
  __bf16* Qh  = (__bf16*)(w + 0 * nElem);
  __bf16* Kh  = (__bf16*)(w + 2 * nElem);
  __bf16* Vh  = (__bf16*)(w + 4 * nElem);
  __bf16* VhT = (__bf16*)(w + 6 * nElem);
  __bf16* Oh  = (__bf16*)(w + 8 * nElem);

  dim3 gGemm(Dm / 128, Mrows / 128);            // (16, 32)
  gemm_nt_wmma<float, float, 0><<<gGemm, 256, 0, stream>>>(q, wq_w, wq_b, Qh, Dm);
  gemm_nt_wmma<float, float, 0><<<gGemm, 256, 0, stream>>>(k, wk_w, wk_b, Kh, Dm);
  gemm_nt_wmma<float, float, 0><<<gGemm, 256, 0, stream>>>(v, wv_w, wv_b, Vh, Dm);
  transpose_v_kernel<<<dim3(Sq / 32, DP / 32, Bsz * Hn), dim3(32, 8), 0, stream>>>(Vh, VhT);
  flash_attn_kernel<<<dim3(Sq / 16, Hn, Bsz), 32, 0, stream>>>(Qh, Kh, VhT, Oh);
  gemm_nt_wmma<__bf16, float, 2><<<gGemm, 256, 0, stream>>>(Oh, wd_w, wd_b, (float*)d_out, Dm);
}
